// RhoWisdomModel_46248207843659
// MI455X (gfx1250) — matrix operand order repaired and checked by reading, not compiled, not run on hardware
//
#include <hip/hip_runtime.h>
#include <math.h>

typedef __attribute__((ext_vector_type(16))) __bf16 v16bf;
typedef __attribute__((ext_vector_type(8)))  __bf16 v8bf;
typedef __attribute__((ext_vector_type(8)))  float  v8f;

// ---------------------------------------------------------------------------
// f32 -> bf16 conversion
// ---------------------------------------------------------------------------
__global__ void cvt_f32_to_bf16(const float* __restrict__ src,
                                __bf16* __restrict__ dst, size_t n) {
    size_t i = (size_t)blockIdx.x * blockDim.x + threadIdx.x;
    if (i < n) dst[i] = (__bf16)src[i];
}

// ---------------------------------------------------------------------------
// Generic WMMA GEMM: C[M,N] = A[M,K] * W[N,K]^T (+bias)(+BN)(+ReLU)
// flags: 1=bias, 2=relu, 4=batchnorm
// 128-thread block = 4 waves (2x2); each wave computes a 32x64 tile
// (2 A-frags x 4 B-frags = 8 WMMAs per K-step). Block tile: 64x128.
// A lane layout (16x32 bf16): lane<16 holds K [k..k+16) of row lr,
// lane>=16 holds K [k+16..k+32). B lane layout (32x16): lane holds
// 16 contiguous K of column lr (lanes 0-15: K lo half, lanes 16-31: K hi half).
// ---------------------------------------------------------------------------
__global__ __launch_bounds__(128)
void gemm_bf16_wmma(const __bf16* __restrict__ A, int lda,
                    const __bf16* __restrict__ W, int ldw,
                    const float* __restrict__ bias,
                    const float* __restrict__ bn_g, const float* __restrict__ bn_b,
                    const float* __restrict__ bn_m, const float* __restrict__ bn_v,
                    float* __restrict__ Cf, __bf16* __restrict__ Cb, int ldc,
                    int M, int N, int K, int flags) {
    const int lane = threadIdx.x & 31;
    const int wave = threadIdx.x >> 5;
    const int half = lane >> 4;      // 0 or 1
    const int lr   = lane & 15;

    const int m0 = (blockIdx.y * 2 + (wave >> 1)) * 32;
    const int n0 = (blockIdx.x * 2 + (wave & 1)) * 64;
    if (m0 >= M || n0 >= N) return;  // wave-uniform exit (EXEC stays all-ones)

    v8f acc[2][4] = {};

    const __bf16* ap0 = A + (size_t)(m0 + lr) * lda + half * 16;
    const __bf16* ap1 = A + (size_t)(m0 + 16 + lr) * lda + half * 16;
    const __bf16* bp0;
    const __bf16* bp1;
    const __bf16* bp2;
    const __bf16* bp3;
    {
        const int n1 = n0 + 16, n2 = n0 + 32, n3 = n0 + 48;
        bp0 = W + (size_t)(n0 + lr) * ldw + half * 16;
        bp1 = W + (size_t)((n1 < N ? n1 : n0) + lr) * ldw + half * 16;
        bp2 = W + (size_t)((n2 < N ? n2 : n0) + lr) * ldw + half * 16;
        bp3 = W + (size_t)((n3 < N ? n3 : n0) + lr) * ldw + half * 16;
    }

    for (int k = 0; k < K; k += 32) {
        union { v16bf v; v8bf h[2]; } a[2], b[4];
        a[0].h[0] = *(const v8bf*)(ap0);
        a[0].h[1] = *(const v8bf*)(ap0 + 8);
        a[1].h[0] = *(const v8bf*)(ap1);
        a[1].h[1] = *(const v8bf*)(ap1 + 8);
        b[0].v = *(const v16bf*)(bp0);
        b[1].v = *(const v16bf*)(bp1);
        b[2].v = *(const v16bf*)(bp2);
        b[3].v = *(const v16bf*)(bp3);

        if (k + 32 < K) {  // next-tile prefetch (global_prefetch_b8)
            __builtin_prefetch(ap0 + 32, 0, 1);
            __builtin_prefetch(ap1 + 32, 0, 1);
            __builtin_prefetch(bp0 + 32, 0, 1);
            __builtin_prefetch(bp1 + 32, 0, 1);
            __builtin_prefetch(bp2 + 32, 0, 1);
            __builtin_prefetch(bp3 + 32, 0, 1);
        }

#pragma unroll
        for (int mi = 0; mi < 2; ++mi)
#pragma unroll
            for (int ni = 0; ni < 4; ++ni)
                acc[mi][ni] = __builtin_amdgcn_wmma_f32_16x16x32_bf16(
                    false, a[mi].v, false, b[ni].v, (short)0, acc[mi][ni],
                    false, false);

        ap0 += 32; ap1 += 32;
        bp0 += 32; bp1 += 32; bp2 += 32; bp3 += 32;
    }

#pragma unroll
    for (int mi = 0; mi < 2; ++mi) {
#pragma unroll
        for (int ni = 0; ni < 4; ++ni) {
            const int nb = n0 + ni * 16;
            if (nb >= N) continue;
            const int col = nb + lr;
            float bia = (flags & 1) ? bias[col] : 0.0f;
            float scale = 1.0f, shift = 0.0f;
            if (flags & 4) {  // BN: g*(x-m)*rsqrt(v+eps)+b = x*scale + shift
                float inv = rsqrtf(bn_v[col] + 1e-5f);
                scale = bn_g[col] * inv;
                shift = bn_b[col] - bn_m[col] * scale;
            }
#pragma unroll
            for (int r = 0; r < 8; ++r) {
                const int rowi = m0 + mi * 16 + half * 8 + r;  // C/D layout
                float x = acc[mi][ni][r] + bia;
                if (flags & 4) x = x * scale + shift;
                if (flags & 2) x = fmaxf(x, 0.0f);
                const size_t o = (size_t)rowi * ldc + col;
                if (Cf) Cf[o] = x;
                if (Cb) Cb[o] = (__bf16)x;
            }
        }
    }
}

// ---------------------------------------------------------------------------
// LayerNorm(D=512) + ReLU, f32 in -> bf16 out (into combined buffer, stride ldo)
// 128 threads per row, 4 elems/thread.
// ---------------------------------------------------------------------------
__global__ __launch_bounds__(128)
void ln_relu_to_bf16(const float* __restrict__ X, const float* __restrict__ g,
                     const float* __restrict__ b, __bf16* __restrict__ out,
                     int ldo, int D) {
    const int row = blockIdx.x;
    const float* x = X + (size_t)row * D;
    float v[4];
    float s = 0.0f;
    for (int i = 0; i < 4; ++i) { v[i] = x[threadIdx.x + i * 128]; s += v[i]; }

    __shared__ float red1[4], red2[4];
    for (int o = 16; o > 0; o >>= 1) s += __shfl_xor(s, o, 32);
    if ((threadIdx.x & 31) == 0) red1[threadIdx.x >> 5] = s;
    __syncthreads();
    const float mu = (red1[0] + red1[1] + red1[2] + red1[3]) / (float)D;

    float q = 0.0f;
    for (int i = 0; i < 4; ++i) { float d = v[i] - mu; q += d * d; }
    for (int o = 16; o > 0; o >>= 1) q += __shfl_xor(q, o, 32);
    if ((threadIdx.x & 31) == 0) red2[threadIdx.x >> 5] = q;
    __syncthreads();
    const float var = (red2[0] + red2[1] + red2[2] + red2[3]) / (float)D;
    const float inv = rsqrtf(var + 1e-5f);

    for (int i = 0; i < 4; ++i) {
        const int c = threadIdx.x + i * 128;
        float y = g[c] * (v[i] - mu) * inv + b[c];
        out[(size_t)row * ldo + c] = (__bf16)fmaxf(y, 0.0f);
    }
}

// ---------------------------------------------------------------------------
// deg[i] = max(#adj[i,:] > 0, 1)
// ---------------------------------------------------------------------------
__global__ void compute_deg(const int* __restrict__ adj, float* __restrict__ deg, int N) {
    const int i = blockIdx.x * blockDim.x + threadIdx.x;
    if (i >= N) return;
    int s = 0;
    for (int j = 0; j < N; ++j) s += (adj[(size_t)i * N + j] > 0) ? 1 : 0;
    deg[i] = fmaxf((float)s, 1.0f);
}

// ---------------------------------------------------------------------------
// GNN masked aggregation:
// Sbar[i,c] = (1/deg[i]) * sum_j mask[i,j] * relu(P[i,c] + E[i,j,c] + Q[j,c] + b1[c])
// block = node i, thread = channel c (C=256).
// ---------------------------------------------------------------------------
__global__ __launch_bounds__(256)
void gnn_aggregate(const float* __restrict__ P, const float* __restrict__ Q,
                   const __bf16* __restrict__ E, const float* __restrict__ b1,
                   const int* __restrict__ adj, const float* __restrict__ deg,
                   __bf16* __restrict__ Sbar, int N, int C) {
    const int i = blockIdx.x;
    const int c = threadIdx.x;
    const float pic = P[(size_t)i * C + c] + b1[c];
    const __bf16* Erow = E + (size_t)i * N * C + c;
    float acc = 0.0f;
    for (int j = 0; j < N; ++j) {
        if (adj[(size_t)i * N + j] > 0) {
            float v = pic + (float)Erow[(size_t)j * C] + Q[(size_t)j * C + c];
            acc += fmaxf(v, 0.0f);
        }
    }
    Sbar[(size_t)i * C + c] = (__bf16)(acc / deg[i]);
}

// ---------------------------------------------------------------------------
// GRU cell update (in place on h): gi/gh are N x 3C (r,z,n gate order)
// ---------------------------------------------------------------------------
__global__ void gru_update(const float* __restrict__ gi, const float* __restrict__ gh,
                           float* __restrict__ hf, __bf16* __restrict__ hb,
                           int N, int C) {
    const int idx = blockIdx.x * blockDim.x + threadIdx.x;
    if (idx >= N * C) return;
    const int row = idx / C, c = idx % C;
    const float* gir = gi + (size_t)row * 3 * C;
    const float* ghr = gh + (size_t)row * 3 * C;
    const float r = 1.0f / (1.0f + expf(-(gir[c] + ghr[c])));
    const float z = 1.0f / (1.0f + expf(-(gir[C + c] + ghr[C + c])));
    const float n = tanhf(gir[2 * C + c] + r * ghr[2 * C + c]);
    const float h = hf[idx];
    const float o = (1.0f - z) * n + z * h;
    hf[idx] = o;
    hb[idx] = (__bf16)o;
}

// ---------------------------------------------------------------------------
// column mean of X (N x C) -> mean[C]
// ---------------------------------------------------------------------------
__global__ void col_mean(const float* __restrict__ X, float* __restrict__ mean,
                         int N, int C) {
    const int c = blockIdx.x * blockDim.x + threadIdx.x;
    if (c >= C) return;
    float s = 0.0f;
    for (int i = 0; i < N; ++i) s += X[(size_t)i * C + c];
    mean[c] = s / (float)N;
}

// ---------------------------------------------------------------------------
// out[row, c] (bf16, stride ldo, pre-offset) = h[row,c] + gmean[c]
// ---------------------------------------------------------------------------
__global__ void add_mean_bf16(const float* __restrict__ h, const float* __restrict__ gmean,
                              __bf16* __restrict__ out, int ldo, int M, int C) {
    const int idx = blockIdx.x * blockDim.x + threadIdx.x;
    if (idx >= M * C) return;
    const int row = idx / C, c = idx % C;
    out[(size_t)row * ldo + c] = (__bf16)(h[(size_t)row * C + c] + gmean[c]);
}

// ---------------------------------------------------------------------------
// Final head: score = clip(sigmoid(out1 . w + b) * calib + calib_b, 0, 1)
// ---------------------------------------------------------------------------
__global__ void head_final(const float* __restrict__ X, const float* __restrict__ w,
                           const float* __restrict__ b, const float* __restrict__ calib,
                           const float* __restrict__ calib_b, float* __restrict__ out,
                           int M, int K) {
    const int i = blockIdx.x * blockDim.x + threadIdx.x;
    if (i >= M) return;
    float s = b[0];
    for (int k = 0; k < K; ++k) s += X[(size_t)i * K + k] * w[k];
    const float raw = 1.0f / (1.0f + expf(-s));
    out[i] = fminf(fmaxf(raw * calib[0] + calib_b[0], 0.0f), 1.0f);
}

// ===========================================================================
extern "C" void kernel_launch(void* const* d_in, const int* in_sizes, int n_in,
                              void* d_out, int out_size, void* d_ws, size_t ws_size,
                              hipStream_t stream) {
    constexpr int B = 8192, FEAT = 512, EMB = 768;
    constexpr int H1 = 512, H2 = 256, H3 = 128;
    constexpr int NN = 256, ED = 32, GH = 256;

    const float* features = (const float*)d_in[0];
    const float* embeddings = (const float*)d_in[1];
    const float* nodes = (const float*)d_in[2];
    const float* edge_feats = (const float*)d_in[3];
    const int*   adjacency = (const int*)d_in[4];
    auto P = [&](int i) { return (const float*)d_in[5 + i]; };
    // params order: 0 fb1_w 1 fb1_b 2 bn1_g 3 bn1_b 4 bn1_m 5 bn1_v 6 fb2_w 7 fb2_b
    // 8 bn2_g 9 bn2_b 10 bn2_m 11 bn2_v 12 attn_in_w 13 attn_in_b 14 attn_out_w
    // 15 attn_out_b 16 ep_w 17 ep_b 18 ln_g 19 ln_b 20 gn_node_w 21 gn_node_b
    // 22 gn_edge_w 23 gn_edge_b 24 gn_m1_w 25 gn_m1_b 26 gn_m2_w 27 gn_m2_b
    // 28 gru_wih 29 gru_whh 30 gru_bih 31 gru_bhh 32 wp1_w 33 wp1_b 34 wbn_g
    // 35 wbn_b 36 wbn_m 37 wbn_v 38 wp2_w 39 wp2_b 40 wp3_w 41 wp3_b 42 out1_w
    // 43 out1_b 44 out2_w 45 out2_b 46 calib 47 calib_b

    char* base = (char*)d_ws;
    size_t off = 0;
    auto alloc = [&](size_t elems, size_t esz) -> void* {
        void* p = base + off;
        off += (elems * esz + 255) & ~(size_t)255;
        return p;
    };
    auto abf = [&](size_t n) { return (__bf16*)alloc(n, 2); };
    auto af  = [&](size_t n) { return (float*)alloc(n, 4); };

    // bf16 weights
    __bf16* fb1w = abf((size_t)H1 * FEAT);
    __bf16* fb2w = abf((size_t)H2 * H1);
    __bf16* wv   = abf((size_t)EMB * EMB);   // attn_in_w rows [2*EMB, 3*EMB)
    __bf16* wo   = abf((size_t)EMB * EMB);
    __bf16* epw  = abf((size_t)H1 * EMB);
    __bf16* gnw  = abf((size_t)GH * GH);
    __bf16* gew  = abf((size_t)GH * ED);
    __bf16* m1w  = abf((size_t)GH * 3 * GH);
    __bf16* m2w  = abf((size_t)GH * GH);
    __bf16* wih  = abf((size_t)3 * GH * GH);
    __bf16* whh  = abf((size_t)3 * GH * GH);
    __bf16* wp1w = abf((size_t)H3 * (H1 + H2));
    __bf16* wp2w = abf((size_t)64 * H3);
    __bf16* wp3w = abf((size_t)32 * 64);
    __bf16* o1w  = abf((size_t)16 * 32);
    // activations
    __bf16* feat_bf = abf((size_t)B * FEAT);
    __bf16* h1_bf   = abf((size_t)B * H1);
    float*  h_f     = af((size_t)B * H2);
    __bf16* emb_bf  = abf((size_t)B * EMB);
    __bf16* v_bf    = abf((size_t)B * EMB);
    __bf16* att_bf  = abf((size_t)B * EMB);
    float*  ep_f    = af((size_t)B * H1);
    __bf16* comb_bf = abf((size_t)B * (H1 + H2));
    __bf16* wft1    = abf((size_t)B * H3);
    __bf16* wft2    = abf((size_t)B * 64);
    __bf16* wft3    = abf((size_t)B * 32);
    float*  o1_f    = af((size_t)B * 16);
    // GNN
    __bf16* nodes_bf = abf((size_t)NN * GH);
    float*  gn_f     = af((size_t)NN * GH);
    __bf16* gn_bf    = abf((size_t)NN * GH);
    __bf16* edge_bf  = abf((size_t)NN * NN * ED);
    __bf16* ged_bf   = abf((size_t)NN * NN * GH);
    __bf16* E1_bf    = abf((size_t)NN * NN * GH);
    float*  Pn = af((size_t)NN * GH);
    float*  Qn = af((size_t)NN * GH);
    __bf16* Sbar = abf((size_t)NN * GH);
    __bf16* msg  = abf((size_t)NN * GH);
    float*  gi_f = af((size_t)NN * 3 * GH);
    float*  gh_f = af((size_t)NN * 3 * GH);
    float*  deg  = af(NN);
    float*  gmean = af(GH);

    auto cvt = [&](const float* s, __bf16* d, size_t n) {
        cvt_f32_to_bf16<<<dim3((unsigned)((n + 255) / 256)), dim3(256), 0, stream>>>(s, d, n);
    };
    auto gemm = [&](const __bf16* A, int lda, const __bf16* W, int ldw,
                    const float* bias, const float* g, const float* bb,
                    const float* m, const float* vv, float* Cf, __bf16* Cb, int ldc,
                    int M, int N, int K, int flags) {
        dim3 grid((unsigned)((N + 127) / 128), (unsigned)((M + 63) / 64));
        gemm_bf16_wmma<<<grid, dim3(128), 0, stream>>>(A, lda, W, ldw, bias, g, bb, m, vv,
                                                       Cf, Cb, ldc, M, N, K, flags);
    };

    // ---- conversions ----
    cvt(features, feat_bf, (size_t)B * FEAT);
    cvt(embeddings, emb_bf, (size_t)B * EMB);
    cvt(nodes, nodes_bf, (size_t)NN * GH);
    cvt(edge_feats, edge_bf, (size_t)NN * NN * ED);
    cvt(P(0), fb1w, (size_t)H1 * FEAT);
    cvt(P(6), fb2w, (size_t)H2 * H1);
    cvt(P(12) + (size_t)2 * EMB * EMB, wv, (size_t)EMB * EMB);  // V slice only
    cvt(P(14), wo, (size_t)EMB * EMB);
    cvt(P(16), epw, (size_t)H1 * EMB);
    cvt(P(20), gnw, (size_t)GH * GH);
    cvt(P(22), gew, (size_t)GH * ED);
    cvt(P(24), m1w, (size_t)GH * 3 * GH);
    cvt(P(26), m2w, (size_t)GH * GH);
    cvt(P(28), wih, (size_t)3 * GH * GH);
    cvt(P(29), whh, (size_t)3 * GH * GH);
    cvt(P(32), wp1w, (size_t)H3 * (H1 + H2));
    cvt(P(38), wp2w, (size_t)64 * H3);
    cvt(P(40), wp3w, (size_t)32 * 64);
    cvt(P(42), o1w, (size_t)16 * 32);

    // ---- feature branch: relu(bn2(lin(relu(bn1(lin(features)))))) ----
    gemm(feat_bf, FEAT, fb1w, FEAT, P(1), P(2), P(3), P(4), P(5),
         nullptr, h1_bf, H1, B, H1, FEAT, 1 | 2 | 4);
    gemm(h1_bf, H1, fb2w, H1, P(7), P(8), P(9), P(10), P(11),
         h_f, nullptr, H2, B, H2, H1, 1 | 2 | 4);

    // ---- attention branch (seq-len 1 => softmax==1 => ctx==v) ----
    gemm(emb_bf, EMB, wv, EMB, P(13) + 2 * EMB, nullptr, nullptr, nullptr, nullptr,
         nullptr, v_bf, EMB, B, EMB, EMB, 1);
    gemm(v_bf, EMB, wo, EMB, P(15), nullptr, nullptr, nullptr, nullptr,
         nullptr, att_bf, EMB, B, EMB, EMB, 1);
    gemm(att_bf, EMB, epw, EMB, P(17), nullptr, nullptr, nullptr, nullptr,
         ep_f, nullptr, H1, B, H1, EMB, 1);
    ln_relu_to_bf16<<<dim3(B), dim3(128), 0, stream>>>(ep_f, P(18), P(19),
                                                       comb_bf, H1 + H2, H1);

    // ---- GNN setup ----
    gemm(nodes_bf, GH, gnw, GH, P(21), nullptr, nullptr, nullptr, nullptr,
         gn_f, gn_bf, GH, NN, GH, GH, 1);
    gemm(edge_bf, ED, gew, ED, P(23), nullptr, nullptr, nullptr, nullptr,
         nullptr, ged_bf, GH, NN * NN, GH, ED, 1);
    // E = gedges @ W1b^T  (iteration-invariant, hoisted out of the loop)
    gemm(ged_bf, GH, m1w + GH, 3 * GH, nullptr, nullptr, nullptr, nullptr, nullptr,
         nullptr, E1_bf, GH, NN * NN, GH, GH, 0);
    compute_deg<<<dim3(1), dim3(NN), 0, stream>>>(adjacency, deg, NN);

    // ---- 3 message-passing iterations ----
    for (int it = 0; it < 3; ++it) {
        gemm(gn_bf, GH, m1w, 3 * GH, nullptr, nullptr, nullptr, nullptr, nullptr,
             Pn, nullptr, GH, NN, GH, GH, 0);                       // src part
        gemm(gn_bf, GH, m1w + 2 * GH, 3 * GH, nullptr, nullptr, nullptr, nullptr,
             nullptr, Qn, nullptr, GH, NN, GH, GH, 0);              // dst part
        gnn_aggregate<<<dim3(NN), dim3(GH), 0, stream>>>(Pn, Qn, E1_bf, P(25),
                                                         adjacency, deg, Sbar, NN, GH);
        // msg = Sbar @ W2^T + b2   (masked-mean commuted through linear m2)
        gemm(Sbar, GH, m2w, GH, P(27), nullptr, nullptr, nullptr, nullptr,
             nullptr, msg, GH, NN, GH, GH, 1);
        gemm(msg, GH, wih, GH, P(30), nullptr, nullptr, nullptr, nullptr,
             gi_f, nullptr, 3 * GH, NN, 3 * GH, GH, 1);
        gemm(gn_bf, GH, whh, GH, P(31), nullptr, nullptr, nullptr, nullptr,
             gh_f, nullptr, 3 * GH, NN, 3 * GH, GH, 1);
        gru_update<<<dim3((NN * GH + 255) / 256), dim3(256), 0, stream>>>(
            gi_f, gh_f, gn_f, gn_bf, NN, GH);
    }

    // ---- combine: feat_out = h + mean(gnodes); combined = [embed_proc | feat_out] ----
    col_mean<<<dim3(1), dim3(GH), 0, stream>>>(gn_f, gmean, NN, GH);
    add_mean_bf16<<<dim3((B * H2 + 255) / 256), dim3(256), 0, stream>>>(
        h_f, gmean, comb_bf + H1, H1 + H2, B, H2);

    // ---- head ----
    gemm(comb_bf, H1 + H2, wp1w, H1 + H2, P(33), P(34), P(35), P(36), P(37),
         nullptr, wft1, H3, B, H3, H1 + H2, 1 | 2 | 4);
    gemm(wft1, H3, wp2w, H3, P(39), nullptr, nullptr, nullptr, nullptr,
         nullptr, wft2, 64, B, 64, H3, 1 | 2);
    gemm(wft2, 64, wp3w, 64, P(41), nullptr, nullptr, nullptr, nullptr,
         nullptr, wft3, 32, B, 32, 64, 1 | 2);
    gemm(wft3, 32, o1w, 32, P(43), nullptr, nullptr, nullptr, nullptr,
         o1_f, nullptr, 16, B, 16, 32, 1 | 2);
    head_final<<<dim3((B + 255) / 256), dim3(256), 0, stream>>>(
        o1_f, P(44), P(45), P(46), P(47), (float*)d_out, B, 16);
}
